// EdgeUpdateLayer_32976758898975
// MI455X (gfx1250) — matrix operand order repaired and checked by reading, not compiled, not run on hardware
//
#include <hip/hip_runtime.h>

// ---------------- CDNA5 / gfx1250 types ----------------
typedef __attribute__((ext_vector_type(16))) __bf16 v16bf;
typedef __attribute__((ext_vector_type(4)))  __bf16 v4bf;
typedef __attribute__((ext_vector_type(8)))  float  v8f;
typedef __attribute__((ext_vector_type(4)))  float  fvec4;

#define DFEAT   256
#define K3      768          // 3*D concatenated K dimension
#define TILE_M  64           // edges per workgroup
#define LDA     776          // bf16 elements per LDS A row (768 + 8 pad), 16B-aligned stride
#define LDX     260          // f32 elements per LDS X row (256 + 4 pad), 16B-aligned stride
#define KSTEPS  24           // 768 / 32
#define NEDGE   500000
#define LRELU_S 0.01f
#define LN_EPS  1e-5f

union Frag32 {                          // 32-byte WMMA operand (16 bf16)
  v16bf  v;
  float4 f4[2];
};

// Repack W (768x256 f32 row-major) into bf16 "B-fragment order":
// dst element g = ((j*KSTEPS + k)*32 + lane)*16 + t
//   col = j*16 + (lane&15), K = k*32 + (t>>3)*16 + (lane>>4)*8 + (t&7)
// so each lane's 32B frag load in the GEMM loop is contiguous & wave-coalesced.
__global__ void wpack_kernel(const float* __restrict__ W,
                             __bf16* __restrict__ W2) {
  int g    = blockIdx.x * blockDim.x + threadIdx.x;   // 0 .. 768*256-1
  int t    = g & 15;
  int lane = (g >> 4) & 31;
  int jk   = g >> 9;
  int k    = jk % KSTEPS;
  int j    = jk / KSTEPS;
  int col  = j * 16 + (lane & 15);
  int kk   = k * 32 + ((t >> 3) << 4) + ((lane >> 4) << 3) + (t & 7);
  W2[g] = (__bf16)W[kk * DFEAT + col];
}

__global__ __launch_bounds__(256, 1)
void edge_update_kernel(const float* __restrict__ H, const float* __restrict__ E,
                        const int* __restrict__ ht, const __bf16* __restrict__ W2,
                        const float* __restrict__ bias, const float* __restrict__ gamma,
                        const float* __restrict__ beta, float* __restrict__ out) {
  extern __shared__ char smem[];
  __bf16* sA = (__bf16*)smem;                                     // [TILE_M][LDA] bf16
  float*  sX = (float*)(smem + (size_t)TILE_M * LDA * 2);         // [TILE_M][LDX] f32

  const int tid  = threadIdx.x;
  const int lane = tid & 31;
  const int wid  = tid >> 5;                       // 0..7
  const long long e0 = (long long)blockIdx.x * TILE_M;

  // ---------- Stage A = [head | E | tail] (64 x 768) as bf16 into LDS ----------
  {
    int row = tid >> 2;                            // 4 threads per edge row
    int sub = tid & 3;
    long long e = e0 + row; if (e >= NEDGE) e = NEDGE - 1;
    int hidx = ht[2 * e];
    int tidx = ht[2 * e + 1];
    const fvec4* ph = (const fvec4*)(H + (size_t)hidx * DFEAT);
    const fvec4* pe = (const fvec4*)(E + (size_t)e    * DFEAT);
    const fvec4* pt = (const fvec4*)(H + (size_t)tidx * DFEAT);
    __bf16* dst = sA + row * LDA;
#pragma unroll
    for (int i = 0; i < 16; ++i) {                 // head -> cols 0..255
      int idx = 4 * i + sub;
      fvec4 f = ph[idx];
      *(v4bf*)(dst + idx * 4) = __builtin_convertvector(f, v4bf);
    }
#pragma unroll
    for (int i = 0; i < 16; ++i) {                 // E -> cols 256..511
      int idx = 4 * i + sub;
      fvec4 f = pe[idx];
      *(v4bf*)(dst + 256 + idx * 4) = __builtin_convertvector(f, v4bf);
    }
#pragma unroll
    for (int i = 0; i < 16; ++i) {                 // tail -> cols 512..767
      int idx = 4 * i + sub;
      fvec4 f = pt[idx];
      *(v4bf*)(dst + 512 + idx * 4) = __builtin_convertvector(f, v4bf);
    }
  }
  __syncthreads();

  // ---------- GEMM: (64 x 768) bf16 @ (768 x 256) bf16 -> f32, via WMMA ----------
  // Waves in a 2(M) x 4(N) grid: each wave owns a 32x64 output tile
  // (2 M-tiles x 4 N-tiles = 8 accumulators), 8 WMMAs per K-step.
  const int wm = wid & 1;                          // 0..1 : rows wm*32 .. +31
  const int wn = wid >> 1;                         // 0..3 : cols wn*64 .. +63
  const __bf16* aRow0 = sA + (wm * 32 + (lane & 15)) * LDA;
  const __bf16* aRow1 = aRow0 + 16 * LDA;
  const int koff = ((lane >> 4) << 3);             // ISA 16-bit A layout: 0 or 8
  const float4* Wf4 = (const float4*)W2;

  v8f acc[2][4];
#pragma unroll
  for (int mt = 0; mt < 2; ++mt)
#pragma unroll
    for (int jj = 0; jj < 4; ++jj) acc[mt][jj] = (v8f){};

#pragma unroll 2
  for (int k = 0; k < KSTEPS; ++k) {
    Frag32 a0, a1, b[4];
    int o = k * 32 + koff;
    a0.f4[0] = *(const float4*)(aRow0 + o);        // K +0..7  (or +8..15)
    a0.f4[1] = *(const float4*)(aRow0 + o + 16);   // K +16..23 (or +24..31)
    a1.f4[0] = *(const float4*)(aRow1 + o);
    a1.f4[1] = *(const float4*)(aRow1 + o + 16);
#pragma unroll
    for (int jj = 0; jj < 4; ++jj) {
      int j = wn * 4 + jj;
      int fi = ((j * KSTEPS + k) * 32 + lane) * 2;
      b[jj].f4[0] = Wf4[fi];
      b[jj].f4[1] = Wf4[fi + 1];
    }
#pragma unroll
    for (int jj = 0; jj < 4; ++jj) {
      acc[0][jj] = __builtin_amdgcn_wmma_f32_16x16x32_bf16(false, a0.v, false, b[jj].v,
                                                           (short)0, acc[0][jj], false, false);
      acc[1][jj] = __builtin_amdgcn_wmma_f32_16x16x32_bf16(false, a1.v, false, b[jj].v,
                                                           (short)0, acc[1][jj], false, false);
    }
  }

  // epilogue: + bias, leaky-ReLU, park in LDS x-buffer
#pragma unroll
  for (int jj = 0; jj < 4; ++jj) {
    int c  = (wn * 4 + jj) * 16 + (lane & 15);
    float bb = bias[c];
#pragma unroll
    for (int mt = 0; mt < 2; ++mt) {
      int mbase = wm * 32 + mt * 16 + ((lane >> 4) << 3);
#pragma unroll
      for (int r = 0; r < 8; ++r) {
        float v = acc[mt][jj][r] + bb;
        v = (v >= 0.f) ? v : LRELU_S * v;
        sX[(mbase + r) * LDX + c] = v;
      }
    }
  }
  __syncthreads();

  // ---------- residual +E and LayerNorm (one wave per 8 rows) ----------
#pragma unroll
  for (int rr = 0; rr < 8; ++rr) {
    int m = wid * 8 + rr;
    long long e  = e0 + m;
    long long ec = (e < NEDGE) ? e : (NEDGE - 1);
    const float* Erow = E + (size_t)ec * DFEAT;
    float x[8];
    float s = 0.f;
#pragma unroll
    for (int i = 0; i < 8; ++i) {
      int c = lane + 32 * i;                       // coalesced
      x[i] = sX[m * LDX + c] + Erow[c];
      s += x[i];
    }
#pragma unroll
    for (int off = 16; off > 0; off >>= 1) s += __shfl_xor(s, off, 32);
    float mu = s * (1.0f / DFEAT);
    float s2 = 0.f;
#pragma unroll
    for (int i = 0; i < 8; ++i) { float d = x[i] - mu; s2 += d * d; }
#pragma unroll
    for (int off = 16; off > 0; off >>= 1) s2 += __shfl_xor(s2, off, 32);
    float rstd = rsqrtf(s2 * (1.0f / DFEAT) + LN_EPS);
    if (e < NEDGE) {
      float* orow = out + (size_t)e * DFEAT;
#pragma unroll
      for (int i = 0; i < 8; ++i) {
        int c = lane + 32 * i;
        orow[c] = gamma[c] * (x[i] - mu) * rstd + beta[c];
      }
    }
  }
}

extern "C" void kernel_launch(void* const* d_in, const int* in_sizes, int n_in,
                              void* d_out, int out_size, void* d_ws, size_t ws_size,
                              hipStream_t stream) {
  const float* H     = (const float*)d_in[0];
  const float* E     = (const float*)d_in[1];
  const int*   ht    = (const int*)d_in[2];
  const float* W     = (const float*)d_in[3];
  const float* bias  = (const float*)d_in[4];
  const float* gamma = (const float*)d_in[5];
  const float* beta  = (const float*)d_in[6];
  float* out = (float*)d_out;
  __bf16* W2 = (__bf16*)d_ws;                      // 768*256 bf16 = 384 KB

  // pack W -> bf16 fragment order (runs every call; deterministic)
  wpack_kernel<<<(K3 * DFEAT) / 256, 256, 0, stream>>>(W, W2);

  int nblk = (NEDGE + TILE_M - 1) / TILE_M;        // 7813
  size_t lds = (size_t)TILE_M * LDA * 2 + (size_t)TILE_M * LDX * 4; // ~166 KB
  edge_update_kernel<<<nblk, 256, lds, stream>>>(H, E, ht, W2, bias, gamma, beta, out);
}
